// GraphSAGE_gravity_15779709845832
// MI455X (gfx1250) — matrix operand order repaired and checked by reading, not compiled, not run on hardware
//
#include <hip/hip_runtime.h>
#include <hip/hip_bf16.h>

typedef __attribute__((ext_vector_type(16))) _Float16 v16h;
typedef __attribute__((ext_vector_type(8)))  _Float16 v8h;
typedef __attribute__((ext_vector_type(8)))  float    v8f;
typedef int v4i_ls __attribute__((vector_size(16)));  // matches builtin param type

#define LRELU(x) ((x) > 0.0f ? (x) : 0.01f * (x))

// ---- CDNA5 async global->LDS copy (ASYNCcnt) with portable fallback --------
#if defined(__has_builtin)
#if __has_builtin(__builtin_amdgcn_global_load_async_to_lds_b128) && \
    __has_builtin(__builtin_amdgcn_s_wait_asynccnt)
#define USE_ASYNC_LDS 1
#endif
#endif

#ifdef USE_ASYNC_LDS
#define ASYNC_CP_B128(g, l)                                       \
  __builtin_amdgcn_global_load_async_to_lds_b128(                 \
      (__attribute__((address_space(1))) v4i_ls*)(g),             \
      (__attribute__((address_space(3))) v4i_ls*)(l), 0, 0)
#define ASYNC_WAIT() __builtin_amdgcn_s_wait_asynccnt(0)
#else
#define ASYNC_CP_B128(g, l) (*(float4*)(l) = *(const float4*)(g))
#define ASYNC_WAIT()
#endif

// ---------------------------------------------------------------- degree
__global__ void deg_kernel(const int* __restrict__ colv, float* __restrict__ cnt, int E) {
  int e = blockIdx.x * blockDim.x + threadIdx.x;
  if (e < E) atomicAdd(&cnt[colv[e]], 1.0f);
}

// ------------------------------------------------- SAGE message scatter-add
// s[col[e], :] += h[row[e], :]   (thread per (edge, 4-float chunk))
__global__ void sage_scatter(const float* __restrict__ h, const int* __restrict__ rowv,
                             const int* __restrict__ colv, float* __restrict__ s,
                             int E, int F) {
  int idx = blockIdx.x * blockDim.x + threadIdx.x;
  int chunks = F >> 2;
  int e = idx / chunks;
  if (e >= E) return;
  int c = (idx - e * chunks) << 2;
  int r = rowv[e], t = colv[e];
  const float4 v = *(const float4*)(h + (size_t)r * F + c);
  float* dst = s + (size_t)t * F + c;
  atomicAdd(dst + 0, v.x);
  atomicAdd(dst + 1, v.y);
  atomicAdd(dst + 2, v.z);
  atomicAdd(dst + 3, v.w);
}

// -------------------------------------------------------- SAGE combine GEMM
// hout = leaky_relu( (s/max(cnt,1)) @ wl + x @ wr + bl )
// One 16-row tile per block (A staged async into LDS); 16-col slab per wave.
__global__ void sage_gemm(const float* __restrict__ s, const float* __restrict__ cnt,
                          const float* __restrict__ xin,
                          const float* __restrict__ wl, const float* __restrict__ bl,
                          const float* __restrict__ wr,
                          float* __restrict__ hout, int Fin, int Fout) {
  __shared__ _Float16 wldsT[64][136];  // B: [n][k] transposed; K = 2*Fin <= 128
  __shared__ float alds[16][132];      // A tile: 16 rows x (2*Fin)
  const int tid = threadIdx.x;
  const int Ktot = 2 * Fin;
  for (int idx = tid; idx < Ktot * Fout; idx += blockDim.x) {
    int k = idx / Fout, n = idx - k * Fout;
    float wv = (k < Fin) ? wl[k * Fout + n] : wr[(k - Fin) * Fout + n];
    wldsT[n][k] = (_Float16)wv;
  }
  {  // async-stage A = [s_row | x_row] per tile row
    const int kq = Ktot >> 2;
    const int m0 = blockIdx.x * 16;
    for (int idx = tid; idx < 16 * kq; idx += blockDim.x) {
      int r = idx / kq, k = (idx - r * kq) << 2;
      const float* gsrc = (k < Fin) ? (s + (size_t)(m0 + r) * Fin + k)
                                    : (xin + (size_t)(m0 + r) * Fin + (k - Fin));
      ASYNC_CP_B128(gsrc, &alds[r][k]);
    }
  }
  ASYNC_WAIT();
  __syncthreads();

  const int lane = tid & 31;
  const int lo = lane & 15, hi = lane >> 4;
  const int n0 = (tid >> 5) * 16;
  const int m = blockIdx.x * 16 + lo;
  const float invc = 1.0f / fmaxf(cnt[m], 1.0f);
  const float* arow = alds[lo];

  v8f acc = {};
  for (int k0 = 0; k0 < Ktot; k0 += 32) {
    const float scale = (k0 < Fin) ? invc : 1.0f;
    float4 q0 = *(const float4*)(arow + k0 + hi * 8);
    float4 q1 = *(const float4*)(arow + k0 + hi * 8 + 4);
    float4 q2 = *(const float4*)(arow + k0 + 16 + hi * 8);
    float4 q3 = *(const float4*)(arow + k0 + 16 + hi * 8 + 4);
    v16h a;
    a[0]  = (_Float16)(q0.x * scale); a[1]  = (_Float16)(q0.y * scale);
    a[2]  = (_Float16)(q0.z * scale); a[3]  = (_Float16)(q0.w * scale);
    a[4]  = (_Float16)(q1.x * scale); a[5]  = (_Float16)(q1.y * scale);
    a[6]  = (_Float16)(q1.z * scale); a[7]  = (_Float16)(q1.w * scale);
    a[8]  = (_Float16)(q2.x * scale); a[9]  = (_Float16)(q2.y * scale);
    a[10] = (_Float16)(q2.z * scale); a[11] = (_Float16)(q2.w * scale);
    a[12] = (_Float16)(q3.x * scale); a[13] = (_Float16)(q3.y * scale);
    a[14] = (_Float16)(q3.z * scale); a[15] = (_Float16)(q3.w * scale);
    v8h b0 = *(const v8h*)&wldsT[n0 + lo][k0 + hi * 8];
    v8h b1 = *(const v8h*)&wldsT[n0 + lo][k0 + 16 + hi * 8];
    v16h b;
#pragma unroll
    for (int i = 0; i < 8; ++i) { b[i] = b0[i]; b[8 + i] = b1[i]; }
    acc = __builtin_amdgcn_wmma_f32_16x16x32_f16(false, a, false, b, (short)0, acc,
                                                 false, false);
  }
  const float bias = bl[n0 + lo];
  float* obase = hout + (size_t)(blockIdx.x * 16) * Fout + n0 + lo;
#pragma unroll
  for (int v = 0; v < 8; ++v) {
    float val = acc[v] + bias;
    obase[(size_t)(hi * 8 + v) * Fout] = LRELU(val);
  }
}

// --------------------------------------------- edge conv: GEMM + scatter-add
// S[col[e], :] += (d[e, :] @ w1)[:]    (bias folded out analytically)
__global__ void edge_gemm_scatter(const float* __restrict__ din,
                                  const float* __restrict__ w1,
                                  const int* __restrict__ colv, float* __restrict__ S,
                                  int Fin, int twoH) {
  __shared__ _Float16 wldsT[128][72];  // B: [n][k]; n = twoH <= 128, k <= 64
  __shared__ float alds[16][68];       // A tile: 16 edges x Fin (<= 64)
  const int tid = threadIdx.x;
  const int Kpad = (Fin + 31) & ~31;
  for (int idx = tid; idx < Kpad * twoH; idx += blockDim.x) {
    int k = idx / twoH, n = idx - k * twoH;
    wldsT[n][k] = (_Float16)((k < Fin) ? w1[k * twoH + n] : 0.0f);
  }
  {  // async-stage A (only the Fin valid floats; pad handled at fragment build)
    const int kq = Fin >> 2;
    const int e0 = blockIdx.x * 16;
    for (int idx = tid; idx < 16 * kq; idx += blockDim.x) {
      int r = idx / kq, k = (idx - r * kq) << 2;
      ASYNC_CP_B128(din + (size_t)(e0 + r) * Fin + k, &alds[r][k]);
    }
  }
  ASYNC_WAIT();
  __syncthreads();

  const int lane = tid & 31;
  const int lo = lane & 15, hi = lane >> 4;
  const int n0 = (tid >> 5) * 16;
  const int e0 = blockIdx.x * 16;
  const float* arow = alds[lo];

  v8f acc = {};
  for (int k0 = 0; k0 < Kpad; k0 += 32) {
    v16h a;
    if (Fin >= 32) {
      float4 q0 = *(const float4*)(arow + k0 + hi * 8);
      float4 q1 = *(const float4*)(arow + k0 + hi * 8 + 4);
      float4 q2 = *(const float4*)(arow + k0 + 16 + hi * 8);
      float4 q3 = *(const float4*)(arow + k0 + 16 + hi * 8 + 4);
      a[0]  = (_Float16)q0.x; a[1]  = (_Float16)q0.y;
      a[2]  = (_Float16)q0.z; a[3]  = (_Float16)q0.w;
      a[4]  = (_Float16)q1.x; a[5]  = (_Float16)q1.y;
      a[6]  = (_Float16)q1.z; a[7]  = (_Float16)q1.w;
      a[8]  = (_Float16)q2.x; a[9]  = (_Float16)q2.y;
      a[10] = (_Float16)q2.z; a[11] = (_Float16)q2.w;
      a[12] = (_Float16)q3.x; a[13] = (_Float16)q3.y;
      a[14] = (_Float16)q3.z; a[15] = (_Float16)q3.w;
    } else {  // Fin == 8 (layer 0): zero-pad K
#pragma unroll
      for (int i = 0; i < 8; ++i) {
        int k  = k0 + hi * 8 + i;
        int k2 = k0 + 16 + hi * 8 + i;
        a[i]     = (_Float16)((k  < Fin) ? arow[k]  : 0.0f);
        a[8 + i] = (_Float16)((k2 < Fin) ? arow[k2] : 0.0f);
      }
    }
    v8h b0 = *(const v8h*)&wldsT[n0 + lo][k0 + hi * 8];
    v8h b1 = *(const v8h*)&wldsT[n0 + lo][k0 + 16 + hi * 8];
    v16h b;
#pragma unroll
    for (int i = 0; i < 8; ++i) { b[i] = b0[i]; b[8 + i] = b1[i]; }
    acc = __builtin_amdgcn_wmma_f32_16x16x32_f16(false, a, false, b, (short)0, acc,
                                                 false, false);
  }
#pragma unroll
  for (int v = 0; v < 8; ++v) {
    int t = colv[e0 + hi * 8 + v];
    atomicAdd(&S[(size_t)t * twoH + n0 + lo], acc[v]);
  }
}

// ---------------------------------------- edge conv: aggregate + update GEMM
// dout = leaky_relu( (S/(deg+1)) @ w2 + (b1 @ w2 + b2) )
__global__ void edge_gemm_out(const float* __restrict__ S, const float* __restrict__ cnt,
                              const float* __restrict__ b1, const float* __restrict__ w2,
                              const float* __restrict__ b2, float* __restrict__ dout,
                              int twoH, int H, int Nn) {
  __shared__ _Float16 wldsT[64][136];  // B: [n][k]; n = H <= 64, k = twoH <= 128
  __shared__ float alds[16][132];      // A tile: 16 rows x twoH
  const int tid = threadIdx.x;
  for (int idx = tid; idx < twoH * H; idx += blockDim.x) {
    int k = idx / H, n = idx - k * H;
    wldsT[n][k] = (_Float16)w2[k * H + n];
  }
  {  // async-stage A
    const int kq = twoH >> 2;
    const int m0 = blockIdx.x * 16;
    for (int idx = tid; idx < 16 * kq; idx += blockDim.x) {
      int r = idx / kq, k = (idx - r * kq) << 2;
      ASYNC_CP_B128(S + (size_t)(m0 + r) * twoH + k, &alds[r][k]);
    }
  }
  ASYNC_WAIT();
  __syncthreads();

  const int lane = tid & 31;
  const int lo = lane & 15, hi = lane >> 4;
  const int n0 = (tid >> 5) * 16;
  const int m = blockIdx.x * 16 + lo;
  const float deg = (m < Nn) ? cnt[m] : 0.0f;
  const float inv = 1.0f / (deg + 1.0f);
  const float* arow = alds[lo];

  v8f acc = {};
  for (int k0 = 0; k0 < twoH; k0 += 32) {
    float4 q0 = *(const float4*)(arow + k0 + hi * 8);
    float4 q1 = *(const float4*)(arow + k0 + hi * 8 + 4);
    float4 q2 = *(const float4*)(arow + k0 + 16 + hi * 8);
    float4 q3 = *(const float4*)(arow + k0 + 16 + hi * 8 + 4);
    v16h a;
    a[0]  = (_Float16)(q0.x * inv); a[1]  = (_Float16)(q0.y * inv);
    a[2]  = (_Float16)(q0.z * inv); a[3]  = (_Float16)(q0.w * inv);
    a[4]  = (_Float16)(q1.x * inv); a[5]  = (_Float16)(q1.y * inv);
    a[6]  = (_Float16)(q1.z * inv); a[7]  = (_Float16)(q1.w * inv);
    a[8]  = (_Float16)(q2.x * inv); a[9]  = (_Float16)(q2.y * inv);
    a[10] = (_Float16)(q2.z * inv); a[11] = (_Float16)(q2.w * inv);
    a[12] = (_Float16)(q3.x * inv); a[13] = (_Float16)(q3.y * inv);
    a[14] = (_Float16)(q3.z * inv); a[15] = (_Float16)(q3.w * inv);
    v8h b0 = *(const v8h*)&wldsT[n0 + lo][k0 + hi * 8];
    v8h b1 = *(const v8h*)&wldsT[n0 + lo][k0 + 16 + hi * 8];
    v16h b;
#pragma unroll
    for (int i = 0; i < 8; ++i) { b[i] = b0[i]; b[8 + i] = b1[i]; }
    acc = __builtin_amdgcn_wmma_f32_16x16x32_f16(false, a, false, b, (short)0, acc,
                                                 false, false);
  }
  // constant row: c0[n] = b2[n] + sum_k b1[k] * w2[k][n]   (f32, exact)
  float c0 = b2[n0 + lo];
  for (int k = 0; k < twoH; ++k) c0 += b1[k] * w2[k * H + n0 + lo];
  float* obase = dout + (size_t)(blockIdx.x * 16) * H + n0 + lo;
#pragma unroll
  for (int v = 0; v < 8; ++v) {
    float val = acc[v] + c0;
    obase[(size_t)(hi * 8 + v) * H] = LRELU(val);
  }
}

// ------------------------------------------------------------- final head
// out[e] = sum_k h[row,k]*h[col,k]*fcw[k] + sum_j d[e,j]*fcw[32+j] + fcb
__global__ void final_kernel(const float* __restrict__ h, const float* __restrict__ dfeat,
                             const int* __restrict__ rowv, const int* __restrict__ colv,
                             const float* __restrict__ fcw, const float* __restrict__ fcb,
                             float* __restrict__ out, int E) {
  int e = blockIdx.x * blockDim.x + threadIdx.x;
  if (e >= E) return;
  int r = rowv[e], c = colv[e];
  const float* hr = h + (size_t)r * 32;
  const float* hc = h + (size_t)c * 32;
  const float* de = dfeat + (size_t)e * 32;
  float acc = fcb[0];
#pragma unroll
  for (int k = 0; k < 32; ++k) acc = fmaf(hr[k] * hc[k], fcw[k], acc);
#pragma unroll
  for (int k = 0; k < 32; ++k) acc = fmaf(de[k], fcw[32 + k], acc);
  out[e] = acc;
}

// ----------------------------------------------------------------- launch
extern "C" void kernel_launch(void* const* d_in, const int* in_sizes, int n_in,
                              void* d_out, int out_size, void* d_ws, size_t ws_size,
                              hipStream_t stream) {
  (void)in_sizes; (void)n_in; (void)out_size;
  const int N = 50000, E = 800000;

  const float* x   = (const float*)d_in[0];
  const float* dis = (const float*)d_in[1];
  const int*   ei  = (const int*)d_in[2];
  const int* rowv = ei;
  const int* colv = ei + E;
  const float *s_wl[4], *s_bl[4], *s_wr[4];
  for (int i = 0; i < 4; ++i) {
    s_wl[i] = (const float*)d_in[3 + 3 * i];
    s_bl[i] = (const float*)d_in[4 + 3 * i];
    s_wr[i] = (const float*)d_in[5 + 3 * i];
  }
  const float *e_w1[4], *e_b1[4], *e_w2[4], *e_b2[4];
  for (int i = 0; i < 4; ++i) {
    e_w1[i] = (const float*)d_in[15 + 4 * i];
    e_b1[i] = (const float*)d_in[16 + 4 * i];
    e_w2[i] = (const float*)d_in[17 + 4 * i];
    e_b2[i] = (const float*)d_in[18 + 4 * i];
  }
  const float* fcw = (const float*)d_in[31];
  const float* fcb = (const float*)d_in[32];
  float* out = (float*)d_out;

  // workspace layout (floats)
  float* ws  = (float*)d_ws;
  float* cnt = ws;                        // N
  float* hA  = ws + 50000;                // N*64
  float* hB  = hA + 3200000;              // N*64
  float* sn  = hB + 3200000;              // N*64 scatter buffer
  float* S   = sn + 3200000;              // E*128 edge scatter buffer
  float* dA  = S + (size_t)E * 128;       // E*64
  float* dB  = dA + (size_t)E * 64;       // E*64
  const size_t need = ((size_t)(dB - ws) + (size_t)E * 64) * sizeof(float);
  if (ws_size < need) return;  // insufficient scratch

  // degree (shared by SAGE mean and edge-conv deg+1)
  (void)hipMemsetAsync(cnt, 0, N * sizeof(float), stream);
  deg_kernel<<<(E + 255) / 256, 256, 0, stream>>>(colv, cnt, E);

  // SAGE chain: x -> hA -> hB -> hA -> hB
  struct SCfg { const float* in; float* outp; int Fin, Fout; };
  SCfg sg[4] = {{x, hA, 64, 64}, {hA, hB, 64, 64}, {hB, hA, 64, 32}, {hA, hB, 32, 32}};
  for (int i = 0; i < 4; ++i) {
    (void)hipMemsetAsync(sn, 0, (size_t)N * sg[i].Fin * sizeof(float), stream);
    int total = E * (sg[i].Fin / 4);
    sage_scatter<<<(total + 255) / 256, 256, 0, stream>>>(sg[i].in, rowv, colv, sn, E,
                                                          sg[i].Fin);
    sage_gemm<<<N / 16, (sg[i].Fout / 16) * 32, 0, stream>>>(
        sn, cnt, sg[i].in, s_wl[i], s_bl[i], s_wr[i], sg[i].outp, sg[i].Fin, sg[i].Fout);
  }

  // Edge chain: dis -> dA -> dB -> dA -> dB
  struct ECfg { const float* in; float* outp; int Fin, H; };
  ECfg eg[4] = {{dis, dA, 8, 64}, {dA, dB, 64, 64}, {dB, dA, 64, 32}, {dA, dB, 32, 32}};
  for (int i = 0; i < 4; ++i) {
    int twoH = 2 * eg[i].H;
    (void)hipMemsetAsync(S, 0, (size_t)E * twoH * sizeof(float), stream);
    edge_gemm_scatter<<<E / 16, (twoH / 16) * 32, 0, stream>>>(eg[i].in, e_w1[i], colv, S,
                                                               eg[i].Fin, twoH);
    edge_gemm_out<<<E / 16, (eg[i].H / 16) * 32, 0, stream>>>(
        S, cnt, e_b1[i], e_w2[i], e_b2[i], eg[i].outp, twoH, eg[i].H, N);
  }

  final_kernel<<<(E + 255) / 256, 256, 0, stream>>>(hB, dB, rowv, colv, fcw, fcb, out, E);
}
